// PosteriorMaskFeature_76596446756981
// MI455X (gfx1250) — matrix operand order repaired and checked by reading, not compiled, not run on hardware
//
#include <hip/hip_runtime.h>
#include <math.h>

typedef __attribute__((ext_vector_type(16))) _Float16 v16h;
typedef __attribute__((ext_vector_type(8)))  float    v8f;

// Fixed problem structure from the reference model
constexpr int HCH   = 64;   // hidden channels
constexpr int MICS  = 8;
constexpr int NPAIR = 4;    // pairs (0,4),(1,5),(2,6),(3,7)
constexpr int FT    = 16;   // freq tile (output)
constexpr int TT    = 32;   // time tile (output)
constexpr int FHALO = FT + 2;
constexpr int THALO = TT + 2;
constexpr float BN_EPS = 1e-5f;

__launch_bounds__(256, 1)
__global__ void pmf_fused_kernel(
    const float* __restrict__ Yr, const float* __restrict__ Yi,
    const float* __restrict__ azi, const float* __restrict__ ele,
    const float* __restrict__ mic_pos, const float* __restrict__ ooc,
    const float* __restrict__ w_mix, const float* __restrict__ b_mix,
    const float* __restrict__ g_mix, const float* __restrict__ be_mix,
    const float* __restrict__ m_mix, const float* __restrict__ v_mix,
    const float* __restrict__ a_mix,
    const float* __restrict__ w_doa, const float* __restrict__ b_doa,
    const float* __restrict__ g_doa, const float* __restrict__ be_doa,
    const float* __restrict__ m_doa, const float* __restrict__ v_doa,
    const float* __restrict__ a_doa,
    const float* __restrict__ w_r1, const float* __restrict__ b_r1,
    const float* __restrict__ g_r, const float* __restrict__ be_r,
    const float* __restrict__ m_r, const float* __restrict__ v_r,
    const float* __restrict__ a_r,
    const float* __restrict__ w_r2, const float* __restrict__ b_r2,
    float* __restrict__ out, int B, int F, int T)
{
    // LDS: total ~159 KB (gfx1250 WGP LDS = 320 KB)
    __shared__ __align__(32) _Float16 sX[FHALO][THALO][HCH]; // interaction tile (f16)  78336 B
    __shared__ __align__(32) _Float16 sW[72][512];           // swizzled A-fragments    73728 B
    __shared__ float sMc[FHALO][HCH];                        // DOA branch per (f,ch)    4608 B
    __shared__ float sDoa[FHALO][8];                         // TPD cos/sin per f-row     576 B
    __shared__ float sZ[FT][TT];                             // final 1x1 accumulator    2048 B

    const int tid = threadIdx.x;
    const int b   = blockIdx.z;
    const int f0  = blockIdx.y * FT;
    const int t0  = blockIdx.x * TT;

    const float amix = a_mix[0];
    const float adoa = a_doa[0];
    const float ar   = a_r[0];

    // ---------------- Phase 0: steering (TPD) features per halo f-row ------------
    if (tid < FHALO) {
        int f = f0 + tid - 1;
        int fc = f < 0 ? 0 : (f >= F ? F - 1 : f);  // clamp; OOB rows become zeros later
        float sa, ca, sE, cE;
        sincosf(azi[b], &sa, &ca);
        sincosf(ele[b], &sE, &cE);
        const float ux = cE * ca, uy = cE * sa, uz = sE;
        const float w = ooc[fc];
        #pragma unroll
        for (int p = 0; p < NPAIR; ++p) {
            float dx = mic_pos[p * 3 + 0] - mic_pos[(p + 4) * 3 + 0];
            float dy = mic_pos[p * 3 + 1] - mic_pos[(p + 4) * 3 + 1];
            float dz = mic_pos[p * 3 + 2] - mic_pos[(p + 4) * 3 + 2];
            float tpd = w * (ux * dx + uy * dy + uz * dz);
            float st, ct;
            sincosf(tpd, &st, &ct);
            sDoa[tid][p]     = ct;
            sDoa[tid][p + 4] = st;
        }
    }
    __syncthreads();

    // ---------------- Phase 1: Mc = PReLU(BN(conv1x1(doa_feat))) per (f-row, ch) --
    for (int idx = tid; idx < FHALO * HCH; idx += 256) {
        int fr = idx >> 6, ch = idx & 63;
        float acc = b_doa[ch];
        #pragma unroll
        for (int c = 0; c < 8; ++c) acc += w_doa[ch * 8 + c] * sDoa[fr][c];
        float s = g_doa[ch] * rsqrtf(v_doa[ch] + BN_EPS);
        float y = acc * s + (be_doa[ch] - m_doa[ch] * s);
        sMc[fr][ch] = y >= 0.f ? y : adoa * y;
    }

    // ---------------- Phase 2: w_r1 fp32 -> f16, pre-swizzled into WMMA A-fragment
// order. Fragment (tap, ob, kb): lane l<16 holds M=ob*16+l, K={0..15}+kb*32;
// lane l>=16 holds M=ob*16+(l-16), K={16..31}+kb*32  (this code's contiguous k order, same as B).
    for (int idx = tid; idx < 72 * 512; idx += 256) {
        int frag = idx >> 9;
        int le   = idx & 511;
        int l = le >> 4, e = le & 15;
        int tap = frag >> 3;
        int rem = frag & 7;
        int ob = rem >> 1, kb = rem & 1;
        int Mrow = ob * 16 + (l & 15);
        int kloc = e + (l >= 16 ? 16 : 0);
        int k = kb * 32 + kloc;
        sW[frag][le] = (_Float16)w_r1[(Mrow * 64 + k) * 9 + tap];
    }

    // zero final accumulator
    for (int idx = tid; idx < FT * TT; idx += 256) ((float*)sZ)[idx] = 0.f;
    __syncthreads();

    // ---------------- Phase 3: interaction = Mx * Mc tile (f16, zero-padded halo) -
    for (int pos = tid; pos < FHALO * THALO; pos += 256) {
        int fr = pos / THALO, tc = pos - fr * THALO;
        int f = f0 + fr - 1, t = t0 + tc - 1;
        if (f < 0 || f >= F || t < 0 || t >= T) {
            #pragma unroll
            for (int ch = 0; ch < HCH; ++ch) sX[fr][tc][ch] = (_Float16)0.f;
        } else {
            // IPD features without atan2: cos/sin of angle difference from complex ratio
            float feat[8];
            const size_t bft = ((size_t)b * MICS * F + f) * (size_t)T + t;
            const size_t mstr = (size_t)F * T;
            #pragma unroll
            for (int p = 0; p < NPAIR; ++p) {
                float ri = Yr[bft + (size_t)p * mstr];
                float ii = Yi[bft + (size_t)p * mstr];
                float rj = Yr[bft + (size_t)(p + 4) * mstr];
                float ij = Yi[bft + (size_t)(p + 4) * mstr];
                float inv = rsqrtf(fmaxf((ri * ri + ii * ii) * (rj * rj + ij * ij), 1e-30f));
                feat[p]     = (ri * rj + ii * ij) * inv;  // cos(ai - aj)
                feat[p + 4] = (ii * rj - ri * ij) * inv;  // sin(ai - aj)
            }
            #pragma unroll 8
            for (int ch = 0; ch < HCH; ++ch) {
                float acc = b_mix[ch];
                #pragma unroll
                for (int c = 0; c < 8; ++c) acc += w_mix[ch * 8 + c] * feat[c];
                float s = g_mix[ch] * rsqrtf(v_mix[ch] + BN_EPS);
                float y = acc * s + (be_mix[ch] - m_mix[ch] * s);
                y = y >= 0.f ? y : amix * y;
                sX[fr][tc][ch] = (_Float16)(y * sMc[fr][ch]);
            }
        }
    }
    __syncthreads();

    // ---------------- Phase 4: 3x3 conv as WMMA GEMM + fused BN/PReLU/1x1 --------
    // 128 tiles: 4 outch-blocks x (16 f-rows x 2 t-halves); 8 waves -> 16 tiles/wave,
    // 18 v_wmma_f32_16x16x32_f16 per tile (9 taps x 2 k-blocks over 64 channels).
    const int wave = tid >> 5;
    const int lane = tid & 31;
    const int lhi  = lane >> 4;    // 0 | 1
    const int ln   = lane & 15;

    for (int tile = wave; tile < 128; tile += 8) {
        int ob    = tile >> 5;
        int pos   = tile & 31;
        int frow  = pos >> 1;
        int tbase = (pos & 1) * 16;

        v8f acc = {};
        #pragma unroll
        for (int tap = 0; tap < 9; ++tap) {
            int df = tap / 3 - 1, dt = tap % 3 - 1;
            // B fragment: lane n<16 -> (N=n, K=0..15), n>=16 -> (N=n-16, K=16..31)
            const _Float16* xrow = &sX[frow + 1 + df][tbase + 1 + dt + ln][lhi * 16];
            #pragma unroll
            for (int kb = 0; kb < 2; ++kb) {
                v16h afrag = *(const v16h*)&sW[(tap * 4 + ob) * 2 + kb][lane * 16];
                v16h bfrag = *(const v16h*)(xrow + kb * 32);
                acc = __builtin_amdgcn_wmma_f32_16x16x32_f16(
                    false, afrag, false, bfrag, (short)0, acc, false, false);
            }
        }

        // Fused epilogue: bias + BN + PReLU + partial w_r2 dot, reduced via LDS atomics.
        // C layout: VGPR r holds M=r (lanes 0-15) / M=r+8 (lanes 16-31), N=lane&15.
        float partial = 0.f;
        #pragma unroll
        for (int r = 0; r < 8; ++r) {
            int ch = ob * 16 + r + lhi * 8;
            float v = acc[r] + b_r1[ch];
            float s = g_r[ch] * rsqrtf(v_r[ch] + BN_EPS);
            float y = v * s + (be_r[ch] - m_r[ch] * s);
            y = y >= 0.f ? y : ar * y;
            partial += y * w_r2[ch];
        }
        atomicAdd(&sZ[frow][tbase + ln], partial);
    }
    __syncthreads();

    // ---------------- Phase 5: sigmoid + store ------------------------------------
    const float br2 = b_r2[0];
    for (int idx = tid; idx < FT * TT; idx += 256) {
        int fr = idx >> 5, tc = idx & 31;
        int f = f0 + fr, t = t0 + tc;
        if (f < F && t < T) {
            float z = sZ[fr][tc] + br2;
            out[((size_t)b * F + f) * (size_t)T + t] = 1.f / (1.f + expf(-z));
        }
    }
}

extern "C" void kernel_launch(void* const* d_in, const int* in_sizes, int n_in,
                              void* d_out, int out_size, void* d_ws, size_t ws_size,
                              hipStream_t stream) {
    (void)n_in; (void)d_ws; (void)ws_size; (void)out_size;
    const float* Yr      = (const float*)d_in[0];
    const float* Yi      = (const float*)d_in[1];
    const float* azi     = (const float*)d_in[2];
    const float* ele     = (const float*)d_in[3];
    const float* mic_pos = (const float*)d_in[4];
    const float* ooc     = (const float*)d_in[5];
    const float* w_mix   = (const float*)d_in[6];
    const float* b_mix   = (const float*)d_in[7];
    const float* g_mix   = (const float*)d_in[8];
    const float* be_mix  = (const float*)d_in[9];
    const float* m_mix   = (const float*)d_in[10];
    const float* v_mix   = (const float*)d_in[11];
    const float* a_mix   = (const float*)d_in[12];
    const float* w_doa   = (const float*)d_in[13];
    const float* b_doa   = (const float*)d_in[14];
    const float* g_doa   = (const float*)d_in[15];
    const float* be_doa  = (const float*)d_in[16];
    const float* m_doa   = (const float*)d_in[17];
    const float* v_doa   = (const float*)d_in[18];
    const float* a_doa   = (const float*)d_in[19];
    const float* w_r1    = (const float*)d_in[20];
    const float* b_r1    = (const float*)d_in[21];
    const float* g_r     = (const float*)d_in[22];
    const float* be_r    = (const float*)d_in[23];
    const float* m_r     = (const float*)d_in[24];
    const float* v_r     = (const float*)d_in[25];
    const float* a_r     = (const float*)d_in[26];
    const float* w_r2    = (const float*)d_in[27];
    const float* b_r2    = (const float*)d_in[28];

    const int B = in_sizes[2];                 // azi has B elements
    const int F = in_sizes[5];                 // omega_over_c has F elements
    const int T = in_sizes[0] / (B * MICS * F);

    dim3 grid((T + TT - 1) / TT, (F + FT - 1) / FT, B);
    pmf_fused_kernel<<<grid, 256, 0, stream>>>(
        Yr, Yi, azi, ele, mic_pos, ooc,
        w_mix, b_mix, g_mix, be_mix, m_mix, v_mix, a_mix,
        w_doa, b_doa, g_doa, be_doa, m_doa, v_doa, a_doa,
        w_r1, b_r1, g_r, be_r, m_r, v_r, a_r, w_r2, b_r2,
        (float*)d_out, B, F, T);
}
